// GraphAttentionLayer_70265664962634
// MI455X (gfx1250) — matrix operand order repaired and checked by reading, not compiled, not run on hardware
//
#include <hip/hip_runtime.h>
#include <hip/hip_bf16.h>
#include <math.h>

typedef __attribute__((ext_vector_type(16))) _Float16 v16h;
typedef __attribute__((ext_vector_type(8)))  _Float16 v8h;
typedef __attribute__((ext_vector_type(8)))  float    v8f;

constexpr int   kN     = 8192;
constexpr int   kInF   = 256;
constexpr int   kHD    = 64;
constexpr float kAlpha = 0.2f;
constexpr float kLog2e = 1.4426950408889634f;

union AZ { int4   v[4]; int   i[16]; };
union FZ { float4 v[4]; float f[16]; };
union AV { v16h   v;    v8h   p[2];  };

// ---------------------------------------------------------------------------
// K1: WhT[h*64+d][n] = (f16) sum_k h[n][k] * W[h][k][d]
// ---------------------------------------------------------------------------
__global__ __launch_bounds__(256) void gat_wh_kernel(
    const float* __restrict__ hin, const float* __restrict__ W,
    _Float16* __restrict__ whT) {
  const int t   = threadIdx.x;
  const int col = t & 127;
  const int hh  = col >> 6;
  const int d   = col & 63;
  const int rg  = t >> 7;
  const int n0  = blockIdx.x * 16 + rg * 8;

  const float* wp = W + (size_t)hh * kInF * kHD + d;
  const float* hp = hin + (size_t)n0 * kInF;

  float acc[8] = {0.f, 0.f, 0.f, 0.f, 0.f, 0.f, 0.f, 0.f};
  for (int k = 0; k < kInF; ++k) {
    float wv = wp[(size_t)k * kHD];
#pragma unroll
    for (int i = 0; i < 8; ++i)
      acc[i] = fmaf(hp[i * kInF + k], wv, acc[i]);
  }
  v8h st;
#pragma unroll
  for (int i = 0; i < 8; ++i) st[i] = (_Float16)acc[i];
  *(v8h*)(whT + (size_t)col * kN + n0) = st;
}

// ---------------------------------------------------------------------------
// K2: f1[h][n] = sum_d WhT[h*64+d][n]*a1[h][d] ; f2 likewise with a2.
// ---------------------------------------------------------------------------
__global__ void gat_f_kernel(const _Float16* __restrict__ whT,
                             const float* __restrict__ a,
                             float* __restrict__ f1, float* __restrict__ f2) {
  const int g  = blockIdx.x * blockDim.x + threadIdx.x;
  const int hh = g >> 13;
  const int n  = g & (kN - 1);
  float s1 = 0.f, s2 = 0.f;
  for (int d = 0; d < kHD; ++d) {
    float w = (float)whT[((size_t)(hh * kHD + d) << 13) + n];
    s1 = fmaf(w, a[hh * 2 * kHD + d],       s1);
    s2 = fmaf(w, a[hh * 2 * kHD + kHD + d], s2);
  }
  f1[g] = s1;
  f2[g] = s2;
}

// ---------------------------------------------------------------------------
// K3: f2max[h] = max_j f2[h][j]
// ---------------------------------------------------------------------------
__global__ void gat_max_kernel(const float* __restrict__ f2,
                               float* __restrict__ f2max) {
  __shared__ float red[256];
  const int hh = blockIdx.x, t = threadIdx.x;
  float m = -3.0e38f;
  for (int j = t; j < kN; j += 256) m = fmaxf(m, f2[hh * kN + j]);
  red[t] = m;
  __syncthreads();
  for (int s = 128; s > 0; s >>= 1) {
    if (t < s) red[t] = fmaxf(red[t], red[t + s]);
    __syncthreads();
  }
  if (t == 0) f2max[hh] = red[0];
}

// ---------------------------------------------------------------------------
// K4: fused masked softmax + attention@Wh (transposed output), WMMA f16.
// One wave: 16 graph rows x both heads over one j-chunk. adj double-buffered.
// ---------------------------------------------------------------------------
template <int C>
__global__ __launch_bounds__(256) void gat_attn_kernel(
    const int* __restrict__ adj, const _Float16* __restrict__ whT,
    const float* __restrict__ f1, const float* __restrict__ f2,
    const float* __restrict__ f2max, float* __restrict__ OT,
    float* __restrict__ Sp) {
  const int lane = threadIdx.x & 31;
  const int li   = lane & 15;
  const int half = lane >> 4;
  const int wave = blockIdx.x * (blockDim.x >> 5) + (threadIdx.x >> 5);
  const int rb   = wave / C;
  const int c    = wave % C;
  const int i0   = rb * 16;
  const int row  = i0 + li;
  const int j0   = c * (kN / C);
  const int jend = j0 + kN / C;
  const int kbA  = half * 16;       // A-layout K base for this lane half
  const int joff = half * 16;       // B-layout j base for this lane half

  float f1r[2], Ml[2];              // Ml = M * log2e (folded softmax shift)
#pragma unroll
  for (int hh = 0; hh < 2; ++hh) {
    f1r[hh] = f1[hh * kN + row];
    float s = f1r[hh] + f2max[hh];
    float M = fmaxf(s, kAlpha * s); // >= all row scores (lrelu monotone)
    Ml[hh]  = M * kLog2e;
  }

  v8f acc[2][4];
#pragma unroll
  for (int hh = 0; hh < 2; ++hh)
#pragma unroll
    for (int dt = 0; dt < 4; ++dt)
#pragma unroll
      for (int r = 0; r < 8; ++r) acc[hh][dt][r] = 0.f;

  float ssum[2] = {0.f, 0.f};
  const int* adjRow = adj + (size_t)row * kN;

  auto tile_body = [&](int jt, const AZ& az) {
#pragma unroll
    for (int hh = 0; hh < 2; ++hh) {
      // Issue all 4 A-tile loads up front; p-computation below hides latency.
      AV av[4];
      const _Float16* abase = whT + ((size_t)(hh * kHD + li) << 13) + jt + kbA;
#pragma unroll
      for (int dt = 0; dt < 4; ++dt) {
        av[dt].p[0] = *(const v8h*)(abase + ((size_t)(dt * 16) << 13));
        av[dt].p[1] = *(const v8h*)(abase + ((size_t)(dt * 16) << 13) + 8);
      }
      FZ fz;
      const float4* fp = (const float4*)(f2 + hh * kN + jt + joff);
      fz.v[0] = fp[0]; fz.v[1] = fp[1]; fz.v[2] = fp[2]; fz.v[3] = fp[3];

      v16h pb;
      float ls[4] = {0.f, 0.f, 0.f, 0.f};
#pragma unroll
      for (int k = 0; k < 16; ++k) {
        float s = f1r[hh] + fz.f[k];
        float e = fmaxf(s, kAlpha * s);                       // leaky_relu
        float q = __builtin_amdgcn_exp2f(fmaf(e, kLog2e, -Ml[hh]));
        float p = (az.i[k] > 0) ? q : 0.f;
        ls[k & 3] += p;
        pb[k] = (_Float16)p;
      }
      ssum[hh] += (ls[0] + ls[1]) + (ls[2] + ls[3]);

#pragma unroll
      for (int dt = 0; dt < 4; ++dt)
        acc[hh][dt] = __builtin_amdgcn_wmma_f32_16x16x32_f16(
            false, av[dt].v, false, pb, (short)0, acc[hh][dt], false, false);
    }
  };

  AZ az0, az1;
  {
    const int4* p = (const int4*)(adjRow + j0 + joff);
    az0.v[0] = p[0]; az0.v[1] = p[1]; az0.v[2] = p[2]; az0.v[3] = p[3];
  }
  for (int jt = j0; jt < jend; jt += 64) {
    {
      const int4* p = (const int4*)(adjRow + jt + 32 + joff);
      az1.v[0] = p[0]; az1.v[1] = p[1]; az1.v[2] = p[2]; az1.v[3] = p[3];
    }
    __builtin_prefetch(adjRow + jt + 512, 0, 0);  // speculative, OOB-safe
    tile_body(jt, az0);
    {
      const int j2 = (jt + 64 < jend) ? jt + 64 : jend - 32;  // clamped reload
      const int4* p = (const int4*)(adjRow + j2 + joff);
      az0.v[0] = p[0]; az0.v[1] = p[1]; az0.v[2] = p[2]; az0.v[3] = p[3];
    }
    tile_body(jt + 32, az1);
  }

#pragma unroll
  for (int hh = 0; hh < 2; ++hh) {
    float tot = ssum[hh] + __shfl_xor(ssum[hh], 16, 32);
    if (half == 0) Sp[((size_t)(c * 2 + hh) << 13) + row] = tot;
#pragma unroll
    for (int dt = 0; dt < 4; ++dt)
#pragma unroll
      for (int r = 0; r < 8; ++r) {
        int d = dt * 16 + r + half * 8;   // C layout: M = r (+8 upper lanes)
        OT[((size_t)((c * 2 + hh) * kHD + d) << 13) + i0 + li] = acc[hh][dt][r];
      }
  }
}

// ---------------------------------------------------------------------------
// K5: combine chunk partials, divide by softmax sum, ELU, interleave heads.
// ---------------------------------------------------------------------------
template <int C>
__global__ void gat_out_kernel(const float* __restrict__ OT,
                               const float* __restrict__ Sp,
                               float* __restrict__ out) {
  const int g   = blockIdx.x * blockDim.x + threadIdx.x;
  const int col = g >> 13;
  const int n   = g & (kN - 1);
  const int hh  = col >> 6;
  const int d   = col & 63;
  float o = 0.f, s = 0.f;
#pragma unroll
  for (int c = 0; c < C; ++c) {
    o += OT[((size_t)((c * 2 + hh) * kHD + d) << 13) + n];
    s += Sp[((size_t)(c * 2 + hh) << 13) + n];
  }
  float v = o / s;
  out[(size_t)n * 128 + col] = v > 0.f ? v : (expf(v) - 1.f);
}

// ---------------------------------------------------------------------------
template <int C>
static void launch_attn(const int* adj, const _Float16* whT, const float* f1,
                        const float* f2, const float* f2m, float* OT, float* Sp,
                        float* out, hipStream_t stream) {
  gat_attn_kernel<C><<<((kN / 16) * C) / 8, 256, 0, stream>>>(
      adj, whT, f1, f2, f2m, OT, Sp);
  gat_out_kernel<C><<<(kN * 128) / 256, 256, 0, stream>>>(OT, Sp, out);
}

extern "C" void kernel_launch(void* const* d_in, const int* in_sizes, int n_in,
                              void* d_out, int out_size, void* d_ws, size_t ws_size,
                              hipStream_t stream) {
  const float* hin = (const float*)d_in[0];
  const int*   adj = (const int*)d_in[1];
  const float* W   = (const float*)d_in[2];
  const float* a   = (const float*)d_in[3];
  float* out = (float*)d_out;

  char* ws = (char*)d_ws;
  size_t off = 0;
  _Float16* whT = (_Float16*)(ws + off); off += (size_t)2 * kHD * kN * sizeof(_Float16);
  float* f1  = (float*)(ws + off); off += (size_t)2 * kN * sizeof(float);
  float* f2  = (float*)(ws + off); off += (size_t)2 * kN * sizeof(float);
  float* f2m = (float*)(ws + off); off += 256;
  float* OT  = (float*)(ws + off);
  const size_t base      = off;
  const size_t per_chunk = (size_t)2 * kHD * kN * sizeof(float)   // OT slice
                         + (size_t)2 * kN * sizeof(float);        // Sp slice

  int C = 1;
  if      (ws_size >= base + 8 * per_chunk) C = 8;
  else if (ws_size >= base + 4 * per_chunk) C = 4;
  else if (ws_size >= base + 2 * per_chunk) C = 2;

  float* Sp = (float*)(ws + base + (size_t)C * 2 * kHD * kN * sizeof(float));

  gat_wh_kernel <<<kN / 16, 256, 0, stream>>>(hin, W, whT);
  gat_f_kernel  <<<(2 * kN) / 256, 256, 0, stream>>>(whT, a, f1, f2);
  gat_max_kernel<<<2, 256, 0, stream>>>(f2, f2m);

  switch (C) {
    case 8: launch_attn<8>(adj, whT, f1, f2, f2m, OT, Sp, out, stream); break;
    case 4: launch_attn<4>(adj, whT, f1, f2, f2m, OT, Sp, out, stream); break;
    case 2: launch_attn<2>(adj, whT, f1, f2, f2m, OT, Sp, out, stream); break;
    default: launch_attn<1>(adj, whT, f1, f2, f2m, OT, Sp, out, stream); break;
  }
}
